// TCMCProbability_19301583029051
// MI455X (gfx1250) — matrix operand order repaired and checked by reading, not compiled, not run on hardware
//
#include <hip/hip_runtime.h>
#include <stdint.h>

#define N_LEAVES 20
#define KINT     19
#define MMOD     8
#define SDIM     4
#define BATCH    200000
#define NNODES   39
#define NEDGE    (2*KINT)        // 38
#define NTILES   (BATCH/16)      // 12500

typedef __attribute__((ext_vector_type(16))) __bf16       v16bf;
typedef __attribute__((ext_vector_type(8)))  float        v8f;

// ---- workspace layout (in 32-bit words) ----
#define AW_WORDS  (NEDGE*2*32*8)   // 19456 : A-operands, bf16-packed, WMMA A-layout
#define PIW_OFF   (AW_WORDS)
#define PIW_WORDS (2*32*8)         // 512   : per-lane pi weights for the root dot
#define PW_OFF    (PIW_OFF + PIW_WORDS)
#define PW_WORDS  (NEDGE*MMOD*16)  // 4864  : P = expm(t*Q), f32

// ---------------- phase 0a: P matrices via expm ----------------
__device__ inline void m4mul(const float* A, const float* B, float* C) {
  #pragma unroll
  for (int i = 0; i < 4; i++)
    #pragma unroll
    for (int j = 0; j < 4; j++) {
      float s = 0.f;
      #pragma unroll
      for (int k = 0; k < 4; k++) s += A[i*4+k] * B[k*4+j];
      C[i*4+j] = s;
    }
}

__device__ inline void compute_pi(const float* pi_inv, int m, float* p) {
  float y0 = pi_inv[m*3+0], y1 = pi_inv[m*3+1], y2 = pi_inv[m*3+2];
  float ns = y0*y0 + y1*y1 + y2*y2;
  float inv = 1.0f / (ns + 1.0f);
  p[0] = 2.f*y0*inv; p[1] = 2.f*y1*inv; p[2] = 2.f*y2*inv; p[3] = (ns - 1.f)*inv;
  #pragma unroll
  for (int c = 0; c < 4; c++) p[c] *= p[c];
}

__global__ void tcmc_phase0a(const float* __restrict__ rates,
                             const float* __restrict__ pi_inv,
                             const float* __restrict__ rho,
                             const float* __restrict__ T,
                             float* __restrict__ Pw) {
  int t = blockIdx.x * blockDim.x + threadIdx.x;
  if (t >= NEDGE * MMOD) return;
  int e = t / MMOD, m = t % MMOD;

  // edge -> (child b, parent a) of the caterpillar tree
  int j = e >> 1;
  int a = N_LEAVES + j;
  int b = (j == 0) ? e : ((e & 1) ? (j + 1) : (N_LEAVES + j - 1));
  float tb = T[b * NNODES + a];

  float p[4]; compute_pi(pi_inv, m, p);

  float r2[6];
  #pragma unroll
  for (int k = 0; k < 6; k++) { float r = rates[m*6+k]; r2[k] = r*r; }
  // triu order: (0,1)(0,2)(0,3)(1,2)(1,3)(2,3)
  float R[16] = {0};
  R[0*4+1]=R[1*4+0]=r2[0]; R[0*4+2]=R[2*4+0]=r2[1]; R[0*4+3]=R[3*4+0]=r2[2];
  R[1*4+2]=R[2*4+1]=r2[3]; R[1*4+3]=R[3*4+1]=r2[4]; R[2*4+3]=R[3*4+2]=r2[5];

  float Q[16];
  #pragma unroll
  for (int i = 0; i < 4; i++) {
    float rs = 0.f;
    #pragma unroll
    for (int jj = 0; jj < 4; jj++) { float v = (i==jj) ? 0.f : R[i*4+jj]*p[jj]; Q[i*4+jj] = v; rs += v; }
    Q[i*4+i] = -rs;
  }
  float emut = 0.f;
  #pragma unroll
  for (int i = 0; i < 4; i++) emut += p[i] * Q[i*4+i];
  emut = fmaxf(-emut, 1e-9f);
  float sc = rho[m] / emut * tb;

  float A[16];
  #pragma unroll
  for (int i = 0; i < 16; i++) A[i] = Q[i] * sc;

  // expm: scaling & squaring + Taylor(12)
  float nrm = 0.f;
  #pragma unroll
  for (int i = 0; i < 4; i++) {
    float rs = 0.f;
    #pragma unroll
    for (int jj = 0; jj < 4; jj++) rs += fabsf(A[i*4+jj]);
    nrm = fmaxf(nrm, rs);
  }
  int s = 0;
  while (nrm > 0.25f && s < 40) { s++; nrm *= 0.5f; }
  float scal = ldexpf(1.0f, -s);
  #pragma unroll
  for (int i = 0; i < 16; i++) A[i] *= scal;

  float E[16], Tm[16], Tn[16];
  #pragma unroll
  for (int i = 0; i < 16; i++) { E[i] = (i%5==0)?1.f:0.f; Tm[i] = E[i]; }
  for (int k = 1; k <= 12; k++) {
    m4mul(Tm, A, Tn);
    float ik = 1.0f / (float)k;
    #pragma unroll
    for (int i = 0; i < 16; i++) { Tm[i] = Tn[i] * ik; E[i] += Tm[i]; }
  }
  for (int it = 0; it < s; it++) {
    m4mul(E, E, Tn);
    #pragma unroll
    for (int i = 0; i < 16; i++) E[i] = Tn[i];
  }
  #pragma unroll
  for (int i = 0; i < 16; i++) Pw[t*16 + i] = E[i];
}

// ---------------- phase 0b: pack WMMA A-operands + pi weights ----------------
__global__ void tcmc_phase0b(const float* __restrict__ Pw,
                             const float* __restrict__ pi_inv,
                             uint32_t* __restrict__ Aw,
                             float* __restrict__ piw) {
  int idx = blockIdx.x * blockDim.x + threadIdx.x;
  if (idx < NEDGE*2*32) {
    int eh = idx >> 5, lane = idx & 31;
    int e = eh >> 1, h = eh & 1;
    int r = lane & 15;                 // A-matrix row M = (m'_loc, c)
    int mp = 4*h + (r >> 2);           // global output model
    int c = r & 3;
    #pragma unroll
    for (int jj = 0; jj < 8; jj++) {
      int kb = ((lane < 16) ? 0 : 16) + 2*jj;
      uint32_t w = 0;
      #pragma unroll
      for (int hh = 0; hh < 2; hh++) {
        int K = kb + hh;                       // K = (m, d)
        int km = K >> 2, kd = K & 3;
        float v = (km == mp) ? Pw[(e*MMOD + mp)*16 + c*4 + kd] : 0.0f;
        uint32_t u = __float_as_uint(v);
        uint32_t bf = (u + 0x7FFFu + ((u >> 16) & 1u)) >> 16;  // RNE f32->bf16
        w |= (bf & 0xFFFFu) << (16*hh);
      }
      Aw[(eh*32 + lane)*8 + jj] = w;
    }
  } else if (idx < NEDGE*2*32 + 64) {
    int k = idx - NEDGE*2*32;
    int th = k >> 5, lane = k & 31;
    int m0 = 4*th + ((lane & 16) ? 2 : 0);
    #pragma unroll
    for (int q = 0; q < 2; q++) {
      float p[4]; compute_pi(pi_inv, m0 + q, p);
      #pragma unroll
      for (int c = 0; c < 4; c++) piw[(th*32 + lane)*8 + 4*q + c] = p[c];
    }
  }
}

// ---------------- main kernel ----------------
__device__ inline v8f wmma8(const uint32_t* Ar, const uint32_t* Br) {
  union U { uint32_t u[8]; v16bf v; } a, b;
  #pragma unroll
  for (int i = 0; i < 8; i++) { a.u[i] = Ar[i]; b.u[i] = Br[i]; }
  v8f c = {};
  return __builtin_amdgcn_wmma_f32_16x16x32_bf16(false, a.v, false, b.v, (short)0, c, false, false);
}

// lane i <-> lane i^16 exchange: one v_permlanex16_b32, no DS traffic
__device__ inline uint32_t swap16(uint32_t v) {
  return (uint32_t)__builtin_amdgcn_permlanex16((int)v, (int)v, 0x76543210, (int)0xfedcba98, false, false);
}

__global__ __launch_bounds__(256) void tcmc_main(const float* __restrict__ X,
                                                 const uint32_t* __restrict__ Aw,
                                                 const float* __restrict__ piw,
                                                 float* __restrict__ out) {
  int wid  = (blockIdx.x * blockDim.x + threadIdx.x) >> 5;
  int lane = threadIdx.x & 31;
  if (wid >= NTILES) return;          // wave-uniform; active waves keep EXEC all-ones
  int site = wid * 16 + (lane & 15);
  bool lolane = (lane & 16) == 0;

  // loop-invariant pi weights for the root dot
  const float4* q0p = (const float4*)(piw + (0*32 + lane)*8);
  const float4* q1p = (const float4*)(piw + (1*32 + lane)*8);
  float4 q0a = q0p[0], q0b = q0p[1], q1a = q1p[0], q1b = q1p[1];

  uint32_t B[8], A[8];

  auto loadA = [&](int eh) {
    const uint4* p = (const uint4*)(Aw + ((eh << 5) + lane)*8);
    uint4 x = p[0], y = p[1];
    A[0]=x.x; A[1]=x.y; A[2]=x.z; A[3]=x.w;
    A[4]=y.x; A[5]=y.y; A[6]=y.z; A[7]=y.w;
  };
  auto leafB = [&](int leaf) {
    const float4 x = *(const float4*)(X + (site*N_LEAVES + leaf)*4);
    uint32_t b0 = __builtin_amdgcn_perm(__float_as_uint(x.y), __float_as_uint(x.x), 0x07060302u);
    uint32_t b1 = __builtin_amdgcn_perm(__float_as_uint(x.w), __float_as_uint(x.z), 0x07060302u);
    B[0]=b0; B[1]=b1; B[2]=b0; B[3]=b1; B[4]=b0; B[5]=b1; B[6]=b0; B[7]=b1;
  };
  // f32 C-tiles (rows=(m,c), cols=site) -> bf16 B-operand: lane-half swap only
  auto conv = [&](const v8f& t0, const v8f& t1) {
    uint32_t u0[4], u1[4];
    #pragma unroll
    for (int jj = 0; jj < 4; jj++) {
      u0[jj] = __builtin_amdgcn_perm(__float_as_uint(t0[2*jj+1]), __float_as_uint(t0[2*jj]), 0x07060302u);
      u1[jj] = __builtin_amdgcn_perm(__float_as_uint(t1[2*jj+1]), __float_as_uint(t1[2*jj]), 0x07060302u);
    }
    #pragma unroll
    for (int jj = 0; jj < 4; jj++) {
      uint32_t s0 = swap16(u0[jj]);
      uint32_t s1 = swap16(u1[jj]);
      B[jj]     = lolane ? u0[jj] : s1;
      B[jj + 4] = lolane ? s0     : u1[jj];
    }
  };

  v8f t0, t1;
  { // node 20: two leaf children (edges 0,1)
    leafB(0);
    loadA(0); v8f d0 = wmma8(A, B);
    loadA(1); v8f d1 = wmma8(A, B);
    leafB(1);
    loadA(2); v8f e0 = wmma8(A, B);
    loadA(3); v8f e1 = wmma8(A, B);
    t0 = d0 * e0; t1 = d1 * e1;
  }
  for (int j = 1; j < KINT; j++) {   // node 20+j: internal child (edge 2j) + leaf j+1 (edge 2j+1)
    conv(t0, t1);
    int ei = 2*j, el = 2*j + 1;
    loadA(ei*2 + 0); v8f d0 = wmma8(A, B);
    loadA(ei*2 + 1); v8f d1 = wmma8(A, B);
    leafB(j + 1);
    loadA(el*2 + 0); v8f e0 = wmma8(A, B);
    loadA(el*2 + 1); v8f e1 = wmma8(A, B);
    t0 = d0 * e0; t1 = d1 * e1;
  }

  // root: out[site, m] = sum_c t[(m,c), site] * pi[m, c]
  float r0 = t0[0]*q0a.x + t0[1]*q0a.y + t0[2]*q0a.z + t0[3]*q0a.w;
  float r1 = t0[4]*q0b.x + t0[5]*q0b.y + t0[6]*q0b.z + t0[7]*q0b.w;
  float r2 = t1[0]*q1a.x + t1[1]*q1a.y + t1[2]*q1a.z + t1[3]*q1a.w;
  float r3 = t1[4]*q1b.x + t1[5]*q1b.y + t1[6]*q1b.z + t1[7]*q1b.w;
  int base = site * MMOD + (lolane ? 0 : 2);
  *(float2*)(out + base)     = make_float2(r0, r1);
  *(float2*)(out + base + 4) = make_float2(r2, r3);
}

// ---------------- launch ----------------
extern "C" void kernel_launch(void* const* d_in, const int* in_sizes, int n_in,
                              void* d_out, int out_size, void* d_ws, size_t ws_size,
                              hipStream_t stream) {
  const float* X      = (const float*)d_in[0];
  const float* rates  = (const float*)d_in[1];
  const float* pi_inv = (const float*)d_in[2];
  const float* rho    = (const float*)d_in[3];
  const float* T      = (const float*)d_in[4];
  float* out = (float*)d_out;

  uint32_t* ws  = (uint32_t*)d_ws;
  uint32_t* Aw  = ws;
  float*    piw = (float*)(ws + PIW_OFF);
  float*    Pw  = (float*)(ws + PW_OFF);

  tcmc_phase0a<<<1, 320, 0, stream>>>(rates, pi_inv, rho, T, Pw);
  tcmc_phase0b<<<(NEDGE*2*32 + 64 + 255)/256, 256, 0, stream>>>(Pw, pi_inv, Aw, piw);
  int blocks = (NTILES + 7) / 8;     // 8 waves per 256-thread block, 1 tile per wave
  tcmc_main<<<blocks, 256, 0, stream>>>(X, Aw, piw, out);
}
